// MultiHeadAttention_41575283425517
// MI455X (gfx1250) — compile-verified, matrix operand order checked by reading, not run on hardware
//
#include <hip/hip_runtime.h>
#include <hip/hip_bf16.h>

// ---- problem constants (from reference) ----
static constexpr int Bc  = 2;
static constexpr int Sc  = 2048;
static constexpr int Dc  = 1024;
static constexpr int Hc  = 16;
static constexpr int DKc = 64;          // Dc / Hc
static constexpr int NT  = Bc * Sc;     // 4096 tokens

typedef __attribute__((ext_vector_type(16))) __bf16 v16bf;
typedef __attribute__((ext_vector_type(8)))  __bf16 v8bf;
typedef __attribute__((ext_vector_type(8)))  float  v8f;

__device__ __forceinline__ v8f wmma_bf16(v16bf a, v16bf b, v8f c) {
  // D = A(16x32 bf16) * B(32x16 bf16) + C(16x16 f32)
  return __builtin_amdgcn_wmma_f32_16x16x32_bf16(
      /*neg_a=*/false, a, /*neg_b=*/false, b,
      /*c_mod=*/(short)0, c, /*reuse_a=*/false, /*reuse_b=*/false);
}

// Load a 16x32 bf16 A-fragment (also used for B when the operand is stored
// transposed, i.e. [N][K] row-major). Per ISA 7.12.2 (16-bit A 16x32):
//   lanes 0-15:  M=lane, K = {0..7} in v[0..7], {16..23} in v[8..15]
//   lanes 16-31: M=lane-16, K = {8..15}, {24..31}
// -> two contiguous 16-byte loads per lane.
__device__ __forceinline__ v16bf load_frag16(const __bf16* __restrict__ base,
                                             int stride, int lane) {
  const int m  = lane & 15;
  const int kh = (lane >> 4) << 3;   // 0 or 8
  const __bf16* p = base + (size_t)m * stride + kh;
  v8bf lo = *(const v8bf*)(p);
  v8bf hi = *(const v8bf*)(p + 16);
  v16bf r;
#pragma unroll
  for (int i = 0; i < 8; ++i) { r[i] = lo[i]; r[i + 8] = hi[i]; }
  return r;
}

// ---- kernel 1: f32 -> bf16 elementwise ----
__global__ __launch_bounds__(256) void cvt_bf16_kernel(
    const float* __restrict__ x, __bf16* __restrict__ y, int n) {
  int i = (blockIdx.x * 256 + threadIdx.x) * 4;
  if (i + 3 < n) {
    float4 f = *(const float4*)(x + i);
    y[i + 0] = (__bf16)f.x;
    y[i + 1] = (__bf16)f.y;
    y[i + 2] = (__bf16)f.z;
    y[i + 3] = (__bf16)f.w;
  }
}

// ---- kernel 2: W[D,D] f32 -> Wt[D,D] bf16 with Wt[n][k] = W[k][n] ----
__global__ __launch_bounds__(256) void wt_bf16_kernel(
    const float* __restrict__ W, __bf16* __restrict__ Wt) {
  __shared__ float tile[32][33];
  const int tx = threadIdx.x & 31;
  const int ty = threadIdx.x >> 5;        // 0..7
  const int k0 = blockIdx.x * 32;
  const int n0 = blockIdx.y * 32;
#pragma unroll
  for (int i = 0; i < 32; i += 8)
    tile[ty + i][tx] = W[(size_t)(k0 + ty + i) * Dc + n0 + tx];
  __syncthreads();
#pragma unroll
  for (int i = 0; i < 32; i += 8)
    Wt[(size_t)(n0 + ty + i) * Dc + k0 + tx] = (__bf16)tile[tx][ty + i];
}

// ---- epilogue helper: scatter one 16x16 C tile per output mode ----
// mode 0: bf16 -> [B*H, S, DK]   (Q)
// mode 1: bf16 -> [B*H, S, DK]   (K)
// mode 2: bf16 -> [B*H, DK, S]   (V transposed)
// mode 3: f32  -> [NT, Dc]       (final output)
__device__ __forceinline__ void store_tile16(v8f acc, int rowBase, int colBase,
                                             const float* __restrict__ bias,
                                             void* __restrict__ out, int mode,
                                             int lane) {
  const int n    = lane & 15;
  const int half = lane >> 4;
  const float bv = bias[colBase + n];
#pragma unroll
  for (int v = 0; v < 8; ++v) {
    const float val = acc[v] + bv;
    const int row = rowBase + v + half * 8;   // token index
    const int col = colBase + n;              // feature index
    if (mode == 3) {
      ((float*)out)[(size_t)row * Dc + col] = val;
    } else {
      const int bidx = row >> 11;             // / S
      const int s    = row & (Sc - 1);
      const int h    = col >> 6;              // / DK
      const int dk   = col & (DKc - 1);
      const __bf16 bw = (__bf16)val;
      if (mode == 2)
        ((__bf16*)out)[(((size_t)(bidx * Hc + h)) * DKc + dk) * Sc + s] = bw;
      else
        ((__bf16*)out)[(((size_t)(bidx * Hc + h)) * Sc + s) * DKc + dk] = bw;
    }
  }
}

// ---- kernel 3/5: Y = A[NT,Dc] * Bt^T + bias ----
// Register-blocked: one wave owns a 32x32 output tile (4 WMMAs / K-step,
// 2 A-frags + 2 B-frags -> 16 FLOP/byte from cache instead of 8).
// Block = 8 waves as 4x2 -> 128x64 block tile.
__global__ __launch_bounds__(256) void proj_gemm_kernel(
    const __bf16* __restrict__ A, const __bf16* __restrict__ Bt,
    const float* __restrict__ bias, void* __restrict__ out, int mode) {
  const int lane    = threadIdx.x & 31;
  const int wave    = threadIdx.x >> 5;
  const int rowBase = blockIdx.x * 128 + (wave >> 1) * 32;
  const int colBase = blockIdx.y * 64  + (wave & 1) * 32;

  const __bf16* aRow0 = A  + (size_t)rowBase * Dc;
  const __bf16* aRow1 = aRow0 + (size_t)16 * Dc;
  const __bf16* bRow0 = Bt + (size_t)colBase * Dc;
  const __bf16* bRow1 = bRow0 + (size_t)16 * Dc;

  v8f acc00 = {}, acc01 = {}, acc10 = {}, acc11 = {};
  for (int k0 = 0; k0 < Dc; k0 += 32) {
    const v16bf a0 = load_frag16(aRow0 + k0, Dc, lane);
    const v16bf a1 = load_frag16(aRow1 + k0, Dc, lane);
    const v16bf b0 = load_frag16(bRow0 + k0, Dc, lane);
    const v16bf b1 = load_frag16(bRow1 + k0, Dc, lane);
    acc00 = wmma_bf16(a0, b0, acc00);
    acc01 = wmma_bf16(a0, b1, acc01);
    acc10 = wmma_bf16(a1, b0, acc10);
    acc11 = wmma_bf16(a1, b1, acc11);
  }

  store_tile16(acc00, rowBase,      colBase,      bias, out, mode, lane);
  store_tile16(acc01, rowBase,      colBase + 16, bias, out, mode, lane);
  store_tile16(acc10, rowBase + 16, colBase,      bias, out, mode, lane);
  store_tile16(acc11, rowBase + 16, colBase + 16, bias, out, mode, lane);
}

// ---- kernel 4: causal flash attention, one wave = 16 query rows ----
__global__ __launch_bounds__(256) void attn_kernel(
    const __bf16* __restrict__ Qb,   // [B*H, S, DK]
    const __bf16* __restrict__ Kb,   // [B*H, S, DK]
    const __bf16* __restrict__ Vt,   // [B*H, DK, S]
    __bf16* __restrict__ Ob) {       // [NT, Dc], col = h*DK + dk
  __shared__ __align__(16) __bf16 Pbuf[8][16][32];   // per-wave P tile

  const int lane  = threadIdx.x & 31;
  const int wave  = threadIdx.x >> 5;
  const int qBase = blockIdx.x * 128 + wave * 16;
  const int bh    = blockIdx.y;
  const int b     = bh >> 4;     // / H
  const int h     = bh & 15;

  const __bf16* Qp = Qb + (size_t)bh * Sc * DKc;
  const __bf16* Kp = Kb + (size_t)bh * Sc * DKc;
  const __bf16* Vp = Vt + (size_t)bh * DKc * Sc;

  const v16bf q0 = load_frag16(Qp + (size_t)qBase * DKc +  0, DKc, lane);
  const v16bf q1 = load_frag16(Qp + (size_t)qBase * DKc + 32, DKc, lane);

  v8f acc0 = {}, acc1 = {}, acc2 = {}, acc3 = {};
  float mrun[8], lrun[8];
#pragma unroll
  for (int v = 0; v < 8; ++v) { mrun[v] = -__builtin_inff(); lrun[v] = 0.f; }

  const int n    = lane & 15;
  const int half = lane >> 4;
  const float scl2 = 0.125f * 1.44269504088896f;  // 1/sqrt(DK) * log2(e)

  // key blocks of 32, up to and including the causal diagonal
  for (int kBase = 0; kBase <= qBase + 15; kBase += 32) {
    // prefetch next key block (K rows + V rows) -> global_prefetch_b8
    if (kBase + 32 <= qBase + 15) {
      __builtin_prefetch(Kp + (size_t)(kBase + 32) * DKc, 0, 0);
      __builtin_prefetch(Vp + kBase + 32, 0, 0);
    }

    // ---- scores: S = Q * K^T (Kb rows are exactly the B^T layout) ----
    const v16bf k00 = load_frag16(Kp + (size_t)(kBase +  0) * DKc +  0, DKc, lane);
    const v16bf k01 = load_frag16(Kp + (size_t)(kBase +  0) * DKc + 32, DKc, lane);
    const v16bf k10 = load_frag16(Kp + (size_t)(kBase + 16) * DKc +  0, DKc, lane);
    const v16bf k11 = load_frag16(Kp + (size_t)(kBase + 16) * DKc + 32, DKc, lane);
    v8f s0 = {}, s1 = {};
    s0 = wmma_bf16(q0, k00, s0);
    s0 = wmma_bf16(q1, k01, s0);
    s1 = wmma_bf16(q0, k10, s1);
    s1 = wmma_bf16(q1, k11, s1);

    // ---- online softmax over the 32-key block (base-2 domain) ----
#pragma unroll
    for (int v = 0; v < 8; ++v) {
      const int qi = qBase + v + half * 8;
      float a = s0[v] * scl2;
      float c = s1[v] * scl2;
      if (kBase + n      > qi) a = -__builtin_inff();
      if (kBase + 16 + n > qi) c = -__builtin_inff();
      float rmax = fmaxf(a, c);
#pragma unroll
      for (int off = 1; off < 16; off <<= 1)
        rmax = fmaxf(rmax, __shfl_xor(rmax, off, 32));
      const float mnew = fmaxf(mrun[v], rmax);
      const float p0 = exp2f(a - mnew);
      const float p1 = exp2f(c - mnew);
      float rsum = p0 + p1;
#pragma unroll
      for (int off = 1; off < 16; off <<= 1)
        rsum += __shfl_xor(rsum, off, 32);
      const float corr = exp2f(mrun[v] - mnew);
      lrun[v] = lrun[v] * corr + rsum;
      mrun[v] = mnew;
      acc0[v] *= corr; acc1[v] *= corr; acc2[v] *= corr; acc3[v] *= corr;
      const int r = v + half * 8;
      Pbuf[wave][r][n]      = (__bf16)p0;   // C-layout -> row-major P tile
      Pbuf[wave][r][n + 16] = (__bf16)p1;
    }
    // LDS per-wave store->load: DS ops are in-order within a wave; no
    // block barrier (trip counts differ per wave).

    // ---- O += P * V  (Vt rows are the B^T layout: contiguous in key) ----
    const v16bf pa  = load_frag16(&Pbuf[wave][0][0], 32, lane);
    const v16bf vb0 = load_frag16(Vp + (size_t) 0 * Sc + kBase, Sc, lane);
    const v16bf vb1 = load_frag16(Vp + (size_t)16 * Sc + kBase, Sc, lane);
    const v16bf vb2 = load_frag16(Vp + (size_t)32 * Sc + kBase, Sc, lane);
    const v16bf vb3 = load_frag16(Vp + (size_t)48 * Sc + kBase, Sc, lane);
    acc0 = wmma_bf16(pa, vb0, acc0);
    acc1 = wmma_bf16(pa, vb1, acc1);
    acc2 = wmma_bf16(pa, vb2, acc2);
    acc3 = wmma_bf16(pa, vb3, acc3);
  }

  // ---- epilogue: divide by l, scatter to [tok, h*DK + dk] ----
#pragma unroll
  for (int v = 0; v < 8; ++v) {
    const int r = v + half * 8;
    const float iv = 1.0f / lrun[v];
    const size_t base =
        ((size_t)(b * Sc + qBase + r)) * Dc + (size_t)h * DKc + n;
    Ob[base +  0] = (__bf16)(acc0[v] * iv);
    Ob[base + 16] = (__bf16)(acc1[v] * iv);
    Ob[base + 32] = (__bf16)(acc2[v] * iv);
    Ob[base + 48] = (__bf16)(acc3[v] * iv);
  }
}

extern "C" void kernel_launch(void* const* d_in, const int* in_sizes, int n_in,
                              void* d_out, int out_size, void* d_ws, size_t ws_size,
                              hipStream_t stream) {
  const float* q  = (const float*)d_in[0];
  const float* k  = (const float*)d_in[1];
  const float* v  = (const float*)d_in[2];
  const float* Wq = (const float*)d_in[3];
  const float* bq = (const float*)d_in[4];
  const float* Wk = (const float*)d_in[5];
  const float* bk = (const float*)d_in[6];
  const float* Wv = (const float*)d_in[7];
  const float* bv = (const float*)d_in[8];
  const float* Wo = (const float*)d_in[9];
  const float* bo = (const float*)d_in[10];
  float* out = (float*)d_out;

  char* ws = (char*)d_ws;
  size_t off = 0;
  auto alloc = [&](size_t bytes) -> void* {
    void* p = ws + off;
    off += (bytes + 255) & ~(size_t)255;
    return p;
  };
  const size_t xBytes = (size_t)NT * Dc * sizeof(__bf16);   // 8 MiB
  const size_t wBytes = (size_t)Dc * Dc * sizeof(__bf16);   // 2 MiB
  __bf16* Xq  = (__bf16*)alloc(xBytes);
  __bf16* Xk  = (__bf16*)alloc(xBytes);
  __bf16* Xv  = (__bf16*)alloc(xBytes);
  __bf16* Wqt = (__bf16*)alloc(wBytes);
  __bf16* Wkt = (__bf16*)alloc(wBytes);
  __bf16* Wvt = (__bf16*)alloc(wBytes);
  __bf16* Wot = (__bf16*)alloc(wBytes);
  __bf16* Qb  = (__bf16*)alloc(xBytes);
  __bf16* Kb  = (__bf16*)alloc(xBytes);
  __bf16* Vt  = (__bf16*)alloc(xBytes);
  __bf16* Ob  = (__bf16*)alloc(xBytes);

  const int nElem = NT * Dc;                 // 4,194,304
  const int cvtBlocks = nElem / (256 * 4);   // 4096

  cvt_bf16_kernel<<<cvtBlocks, 256, 0, stream>>>(q, Xq, nElem);
  cvt_bf16_kernel<<<cvtBlocks, 256, 0, stream>>>(k, Xk, nElem);
  cvt_bf16_kernel<<<cvtBlocks, 256, 0, stream>>>(v, Xv, nElem);

  dim3 tGrid(Dc / 32, Dc / 32);
  wt_bf16_kernel<<<tGrid, 256, 0, stream>>>(Wq, Wqt);
  wt_bf16_kernel<<<tGrid, 256, 0, stream>>>(Wk, Wkt);
  wt_bf16_kernel<<<tGrid, 256, 0, stream>>>(Wv, Wvt);
  wt_bf16_kernel<<<tGrid, 256, 0, stream>>>(Wo, Wot);

  dim3 gGrid(NT / 128, Dc / 64);             // 32 x 16
  proj_gemm_kernel<<<gGrid, 256, 0, stream>>>(Xq, Wqt, bq, Qb, 0);
  proj_gemm_kernel<<<gGrid, 256, 0, stream>>>(Xk, Wkt, bk, Kb, 1);
  proj_gemm_kernel<<<gGrid, 256, 0, stream>>>(Xv, Wvt, bv, Vt, 2);

  dim3 aGrid(Sc / 128, Bc * Hc);             // 16 x 32
  attn_kernel<<<aGrid, 256, 0, stream>>>(Qb, Kb, Vt, Ob);

  proj_gemm_kernel<<<gGrid, 256, 0, stream>>>(Ob, Wot, bo, out, 3);
}
